// MoELayer_4440996184493
// MI455X (gfx1250) — matrix operand order repaired and checked by reading, not compiled, not run on hardware
//
#include <hip/hip_runtime.h>
#include <stdint.h>

// ---------------- problem constants ----------------
constexpr int T = 8192;
constexpr int D = 1024;
constexpr int F = 4096;
constexpr int E = 8;

// ---------------- tile config ----------------
constexpr int BM = 128;   // token rows per block
constexpr int BN = 64;    // output cols per block
constexpr int BK = 32;    // K depth per iteration (matches wmma 16x16x32)

typedef __attribute__((ext_vector_type(16))) __bf16 v16bf;
typedef __attribute__((ext_vector_type(8)))  __bf16 v8bf;
typedef __attribute__((ext_vector_type(2)))  __bf16 v2bf;
typedef __attribute__((ext_vector_type(8)))  float  v8f;
typedef __attribute__((ext_vector_type(4)))  int    v4i;

typedef __attribute__((address_space(1))) v4i* global_v4i_ptr;
typedef __attribute__((address_space(3))) v4i* lds_v4i_ptr;

struct FragBF { union { v16bf v; v8bf h[2]; }; };

#if __has_builtin(__builtin_amdgcn_global_load_async_to_lds_b128)
#define USE_ASYNC_LDS 1
#endif

__device__ __forceinline__ uint32_t bf_bits(float f) {
  uint32_t u = __builtin_bit_cast(uint32_t, f);
  u += 0x7FFFu + ((u >> 16) & 1u);           // round-to-nearest-even
  return u >> 16;
}

__device__ __forceinline__ __bf16 f2bf(float f) {
  uint16_t hv = (uint16_t)bf_bits(f);
  return __builtin_bit_cast(__bf16, hv);
}

__device__ __forceinline__ uint32_t pack_bf16x2(float a, float b) {
#if __has_builtin(__builtin_amdgcn_cvt_pk_bf16_f32)
  v2bf r = __builtin_amdgcn_cvt_pk_bf16_f32(a, b);
  return __builtin_bit_cast(uint32_t, r);
#else
  return (bf_bits(b) << 16) | bf_bits(a);
#endif
}

#if defined(USE_ASYNC_LDS)
__device__ __forceinline__ void async_copy_b128(const void* gsrc, void* ldst) {
  __builtin_amdgcn_global_load_async_to_lds_b128(
      (global_v4i_ptr)const_cast<void*>(gsrc),
      (lds_v4i_ptr)ldst, 0, 0);
}
#else
__device__ __forceinline__ void async_copy_b128(const void* gsrc, void* ldst) {
  *(uint4*)ldst = *(const uint4*)gsrc;       // sync fallback
}
#endif

__device__ __forceinline__ void wait_async_lds() {
#if defined(USE_ASYNC_LDS)
#if __has_builtin(__builtin_amdgcn_s_wait_asynccnt)
  __builtin_amdgcn_s_wait_asynccnt(0);
#else
  asm volatile("s_wait_asynccnt 0x0" ::: "memory");
#endif
#endif
}

// ---------------- 0) one-shot f32 -> bf16 conversion (memory-bound pre-pass) ----------------
__global__ __launch_bounds__(256) void cvt_f32_bf16(
    const float* __restrict__ src, __bf16* __restrict__ dst, size_t n8)
{
  const size_t i = (size_t)blockIdx.x * 256 + threadIdx.x;
  if (i >= n8) return;
  const float4* s = (const float4*)(src + i * 8);
  const float4 f0 = s[0], f1 = s[1];
  uint4 u;
  u.x = pack_bf16x2(f0.x, f0.y);
  u.y = pack_bf16x2(f0.z, f0.w);
  u.z = pack_bf16x2(f1.x, f1.y);
  u.w = pack_bf16x2(f1.z, f1.w);
  *(uint4*)(dst + i * 8) = u;
}

// ---------------- 1) router: top-2 + softmax, per-expert counts ----------------
__global__ __launch_bounds__(256) void moe_router(
    const float* __restrict__ x, const float* __restrict__ gw,
    int* __restrict__ counts, int* __restrict__ sel, float* __restrict__ wgt)
{
  const int wid  = threadIdx.x >> 5;
  const int lane = threadIdx.x & 31;
  const int t = blockIdx.x * 8 + wid;
  if (t >= T) return;

  float acc[E] = {};
  for (int d = lane; d < D; d += 32) {
    const float xv = x[(size_t)t * D + d];
    const float4* g = (const float4*)(gw + (size_t)d * E);
    const float4 g0 = g[0], g1 = g[1];
    acc[0] += xv * g0.x; acc[1] += xv * g0.y; acc[2] += xv * g0.z; acc[3] += xv * g0.w;
    acc[4] += xv * g1.x; acc[5] += xv * g1.y; acc[6] += xv * g1.z; acc[7] += xv * g1.w;
  }
#pragma unroll
  for (int off = 16; off > 0; off >>= 1)
#pragma unroll
    for (int e = 0; e < E; ++e) acc[e] += __shfl_xor(acc[e], off);

  if (lane == 0) {
    int e1 = 0; float a1 = acc[0];
#pragma unroll
    for (int e = 1; e < E; ++e) if (acc[e] > a1) { a1 = acc[e]; e1 = e; }
    int e2 = -1; float a2 = -3.4e38f;
#pragma unroll
    for (int e = 0; e < E; ++e) if (e != e1 && acc[e] > a2) { a2 = acc[e]; e2 = e; }
    const float p = __expf(a2 - a1);
    const float s = 1.0f + p;
    sel[t * 2 + 0] = e1; sel[t * 2 + 1] = e2;
    wgt[t * 2 + 0] = 1.0f / s; wgt[t * 2 + 1] = p / s;
    atomicAdd(&counts[e1], 1);
    atomicAdd(&counts[e2], 1);
  }
}

// ---------------- 2) exclusive scan over 8 experts ----------------
__global__ void moe_scan(const int* __restrict__ counts, int* __restrict__ offsets)
{
  int run = 0;
  for (int e = 0; e < E; ++e) { offsets[e] = run; run += counts[e]; }
  offsets[E] = run;
}

// ---------------- 3) scatter tokens into expert buckets ----------------
__global__ __launch_bounds__(256) void moe_fill(
    const int* __restrict__ sel, const float* __restrict__ wgt,
    const int* __restrict__ offsets, int* __restrict__ fillc,
    int* __restrict__ btok, float* __restrict__ bw)
{
  const int t = blockIdx.x * 256 + threadIdx.x;
  if (t >= T) return;
#pragma unroll
  for (int k = 0; k < 2; ++k) {
    const int e = sel[t * 2 + k];
    const int pos = offsets[e] + atomicAdd(&fillc[e], 1);
    btok[pos] = t;
    bw[pos] = wgt[t * 2 + k];
  }
}

// ---------------- 4) up-projection: H = silu(Xg@w1) * (Xg@w3), bf16 out ----------------
__global__ __launch_bounds__(256) void moe_up(
    const __bf16* __restrict__ xb,
    const __bf16* __restrict__ Wb1, const __bf16* __restrict__ Wb3,
    const int* __restrict__ counts, const int* __restrict__ offsets,
    const int* __restrict__ btok, __bf16* __restrict__ H)
{
  const int e   = blockIdx.z;
  const int cnt = counts[e];
  const int m0  = blockIdx.y * BM;
  if (m0 >= cnt) return;                 // uniform early exit
  const int n0   = blockIdx.x * BN;
  const int base = offsets[e];

  __shared__ __bf16 Xs [BM][BK + 8];
  __shared__ __bf16 W1s[BK][BN + 8];
  __shared__ __bf16 W3s[BK][BN + 8];

  const int tid   = threadIdx.x;
  const int lrow  = tid >> 1;            // 0..127, X-tile row
  const int xc    = (tid & 1) * 16;      // 16 cols per thread
  const int wrow  = tid >> 3;            // 0..31, W-tile row
  const int wc    = (tid & 7) * 8;       // 8 cols per thread

  const bool rowok = (m0 + lrow) < cnt;
  int tok = 0;
  if (rowok) tok = btok[base + m0 + lrow];

  // padding rows stay zero for the whole k-loop
  if (!rowok) {
    *(uint4*)&Xs[lrow][xc]     = make_uint4(0u, 0u, 0u, 0u);
    *(uint4*)&Xs[lrow][xc + 8] = make_uint4(0u, 0u, 0u, 0u);
  }

  const int wid  = tid >> 5;             // 0..7 waves
  const int lane = tid & 31;
  const int wm   = wid >> 1;             // 0..3  (32 rows each)
  const int wn   = wid & 1;              // 0..1  (32 cols each)

  v8f acc1[2][2] = {};
  v8f acc3[2][2] = {};

  for (int k0 = 0; k0 < D; k0 += BK) {
    // ---- stage X tile (bf16, gathered rows; pure global->LDS DMA) ----
    if (rowok) {
      const __bf16* gsrc = xb + (size_t)tok * D + k0 + xc;
      async_copy_b128(gsrc,     &Xs[lrow][xc]);
      async_copy_b128(gsrc + 8, &Xs[lrow][xc + 8]);
    }
    // ---- stage W1 / W3 tiles (bf16; pure global->LDS DMA) ----
    {
      const size_t wb = ((size_t)e * D + (k0 + wrow)) * F + n0 + wc;
      async_copy_b128(Wb1 + wb, &W1s[wrow][wc]);
      async_copy_b128(Wb3 + wb, &W3s[wrow][wc]);
    }
    wait_async_lds();
    __syncthreads();

    // ---- fragments ----
    FragBF a[2], b1f[2], b3f[2];
    const int koff = (lane & 16) ? 8 : 0;
#pragma unroll
    for (int tm = 0; tm < 2; ++tm) {
      const int arow = wm * 32 + tm * 16 + (lane & 15);
      a[tm].h[0] = *(const v8bf*)&Xs[arow][koff];
      a[tm].h[1] = *(const v8bf*)&Xs[arow][koff + 16];
    }
#pragma unroll
    for (int tn = 0; tn < 2; ++tn) {
      const int col = wn * 32 + tn * 16 + (lane & 15);
#pragma unroll
      for (int s = 0; s < 16; ++s) {
        const int kk = (s < 8) ? (koff + s) : (16 + koff + (s - 8));
        b1f[tn].v[s] = W1s[kk][col];
        b3f[tn].v[s] = W3s[kk][col];
      }
    }
#pragma unroll
    for (int tm = 0; tm < 2; ++tm)
#pragma unroll
      for (int tn = 0; tn < 2; ++tn) {
        acc1[tm][tn] = __builtin_amdgcn_wmma_f32_16x16x32_bf16(
            false, a[tm].v, false, b1f[tn].v, (short)0, acc1[tm][tn], false, false);
        acc3[tm][tn] = __builtin_amdgcn_wmma_f32_16x16x32_bf16(
            false, a[tm].v, false, b3f[tn].v, (short)0, acc3[tm][tn], false, false);
      }
    __syncthreads();
  }

  // ---- fused SwiGLU epilogue, store bf16 H ----
#pragma unroll
  for (int tm = 0; tm < 2; ++tm)
#pragma unroll
    for (int j = 0; j < 8; ++j) {
      const int m  = wm * 32 + tm * 16 + j + ((lane & 16) ? 8 : 0);
      const int gr = m0 + m;
      if (gr < cnt) {
#pragma unroll
        for (int tn = 0; tn < 2; ++tn) {
          const float s1 = acc1[tm][tn][j];
          const float s3 = acc3[tm][tn][j];
          const float hval = s1 * (1.0f / (1.0f + __expf(-s1))) * s3;
          const int n = n0 + wn * 32 + tn * 16 + (lane & 15);
          H[(size_t)(base + gr) * F + n] = f2bf(hval);
        }
      }
    }
}

// ---------------- 5) down-projection + weighted scatter-add ----------------
__global__ __launch_bounds__(256) void moe_down(
    const __bf16* __restrict__ H, const __bf16* __restrict__ Wb2,
    const int* __restrict__ counts, const int* __restrict__ offsets,
    const int* __restrict__ btok, const float* __restrict__ bw,
    float* __restrict__ out)
{
  const int e   = blockIdx.z;
  const int cnt = counts[e];
  const int m0  = blockIdx.y * BM;
  if (m0 >= cnt) return;
  const int n0   = blockIdx.x * BN;
  const int base = offsets[e];

  __shared__ __bf16 Hs [BM][BK + 8];
  __shared__ __bf16 W2s[BK][BN + 8];

  const int tid   = threadIdx.x;
  const int lrow  = tid >> 1;
  const int hc    = (tid & 1) * 16;
  const int wrow  = tid >> 3;
  const int wc    = (tid & 7) * 8;

  const bool rowok = (m0 + lrow) < cnt;

  // padding rows stay zero for the whole k-loop
  if (!rowok) {
    *(uint4*)&Hs[lrow][hc]     = make_uint4(0u, 0u, 0u, 0u);
    *(uint4*)&Hs[lrow][hc + 8] = make_uint4(0u, 0u, 0u, 0u);
  }

  const int wid  = tid >> 5;
  const int lane = tid & 31;
  const int wm   = wid >> 1;
  const int wn   = wid & 1;

  v8f acc[2][2] = {};

  for (int k0 = 0; k0 < F; k0 += BK) {
    // ---- stage H tile (bf16; pure global->LDS DMA) ----
    if (rowok) {
      const __bf16* gsrc = H + (size_t)(base + m0 + lrow) * F + k0 + hc;
      async_copy_b128(gsrc,     &Hs[lrow][hc]);
      async_copy_b128(gsrc + 8, &Hs[lrow][hc + 8]);
    }
    // ---- stage W2 tile (bf16; pure global->LDS DMA) ----
    {
      const size_t wb = ((size_t)e * F + (k0 + wrow)) * D + n0 + wc;
      async_copy_b128(Wb2 + wb, &W2s[wrow][wc]);
    }
    wait_async_lds();
    __syncthreads();

    FragBF a[2], bf[2];
    const int koff = (lane & 16) ? 8 : 0;
#pragma unroll
    for (int tm = 0; tm < 2; ++tm) {
      const int arow = wm * 32 + tm * 16 + (lane & 15);
      a[tm].h[0] = *(const v8bf*)&Hs[arow][koff];
      a[tm].h[1] = *(const v8bf*)&Hs[arow][koff + 16];
    }
#pragma unroll
    for (int tn = 0; tn < 2; ++tn) {
      const int col = wn * 32 + tn * 16 + (lane & 15);
#pragma unroll
      for (int s = 0; s < 16; ++s) {
        const int kk = (s < 8) ? (koff + s) : (16 + koff + (s - 8));
        bf[tn].v[s] = W2s[kk][col];
      }
    }
#pragma unroll
    for (int tm = 0; tm < 2; ++tm)
#pragma unroll
      for (int tn = 0; tn < 2; ++tn)
        acc[tm][tn] = __builtin_amdgcn_wmma_f32_16x16x32_bf16(
            false, a[tm].v, false, bf[tn].v, (short)0, acc[tm][tn], false, false);
    __syncthreads();
  }

  // ---- weighted scatter-add into out ----
#pragma unroll
  for (int tm = 0; tm < 2; ++tm)
#pragma unroll
    for (int j = 0; j < 8; ++j) {
      const int m  = wm * 32 + tm * 16 + j + ((lane & 16) ? 8 : 0);
      const int gr = m0 + m;
      if (gr < cnt) {
        const int   tok = btok[base + gr];
        const float w   = bw[base + gr];
#pragma unroll
        for (int tn = 0; tn < 2; ++tn) {
          const int n = n0 + wn * 32 + tn * 16 + (lane & 15);
          atomicAdd(&out[(size_t)tok * D + n], w * acc[tm][tn][j]);
        }
      }
    }
}

// ---------------- host-side launch ----------------
extern "C" void kernel_launch(void* const* d_in, const int* in_sizes, int n_in,
                              void* d_out, int out_size, void* d_ws, size_t ws_size,
                              hipStream_t stream) {
  const float* x  = (const float*)d_in[0];
  const float* gw = (const float*)d_in[1];
  const float* w1 = (const float*)d_in[2];
  const float* w2 = (const float*)d_in[3];
  const float* w3 = (const float*)d_in[4];
  float* out = (float*)d_out;

  // workspace layout (bytes)
  constexpr size_t OFF_META = 0;                         // counts/fill/offsets
  constexpr size_t OFF_SEL  = 128;
  constexpr size_t OFF_WGT  = OFF_SEL  + (size_t)T * 2 * 4;
  constexpr size_t OFF_BTOK = OFF_WGT  + (size_t)T * 2 * 4;
  constexpr size_t OFF_BW   = OFF_BTOK + (size_t)T * 2 * 4;
  constexpr size_t OFF_XB   = 262400;                    // T*D bf16
  constexpr size_t OFF_W1B  = OFF_XB  + (size_t)T * D * 2;
  constexpr size_t OFF_W3B  = OFF_W1B + (size_t)E * D * F * 2;
  constexpr size_t OFF_W2B  = OFF_W3B + (size_t)E * D * F * 2;
  constexpr size_t OFF_H    = OFF_W2B + (size_t)E * F * D * 2;

  char* ws = (char*)d_ws;
  int*    counts  = (int*)(ws + OFF_META);
  int*    fillc   = (int*)(ws + OFF_META + 32);
  int*    offsets = (int*)(ws + OFF_META + 64);
  int*    sel     = (int*)(ws + OFF_SEL);
  float*  wgt     = (float*)(ws + OFF_WGT);
  int*    btok    = (int*)(ws + OFF_BTOK);
  float*  bw      = (float*)(ws + OFF_BW);
  __bf16* xb      = (__bf16*)(ws + OFF_XB);
  __bf16* w1b     = (__bf16*)(ws + OFF_W1B);
  __bf16* w3b     = (__bf16*)(ws + OFF_W3B);
  __bf16* w2b     = (__bf16*)(ws + OFF_W2B);
  __bf16* H       = (__bf16*)(ws + OFF_H);

  (void)hipMemsetAsync(ws, 0, 128, stream);                          // counts/fill
  (void)hipMemsetAsync(d_out, 0, (size_t)T * D * sizeof(float), stream);

  // one-shot bf16 pre-conversion (memory-bound)
  const size_t nx = (size_t)T * D / 8;         // 1,048,576 groups
  const size_t nw = (size_t)E * D * F / 8;     // 4,194,304 groups
  cvt_f32_bf16<<<(unsigned)(nx / 256), 256, 0, stream>>>(x,  xb,  nx);
  cvt_f32_bf16<<<(unsigned)(nw / 256), 256, 0, stream>>>(w1, w1b, nw);
  cvt_f32_bf16<<<(unsigned)(nw / 256), 256, 0, stream>>>(w3, w3b, nw);
  cvt_f32_bf16<<<(unsigned)(nw / 256), 256, 0, stream>>>(w2, w2b, nw);

  // routing
  moe_router<<<T / 8, 256, 0, stream>>>(x, gw, counts, sel, wgt);
  moe_scan  <<<1, 1, 0, stream>>>(counts, offsets);
  moe_fill  <<<(T + 255) / 256, 256, 0, stream>>>(sel, wgt, offsets, fillc, btok, bw);

  // expert GEMMs
  moe_up  <<<dim3(F / BN, T / BM, E), 256, 0, stream>>>(xb, w1b, w3b, counts, offsets, btok, H);
  moe_down<<<dim3(D / BN, T / BM, E), 256, 0, stream>>>(H, w2b, counts, offsets, btok, bw, out);
}
